// RecurrentModel_3504693314113
// MI455X (gfx1250) — compile-verified, matrix operand order checked by reading, not run on hardware
//
#include <hip/hip_runtime.h>
#include <math.h>

typedef __attribute__((ext_vector_type(16))) __bf16 v16bf;
typedef __attribute__((ext_vector_type(8)))  __bf16 v8bf;
typedef __attribute__((ext_vector_type(8)))  float  v8f;

#define WMMA_BF16(a, b, c) \
    __builtin_amdgcn_wmma_f32_16x16x32_bf16(false, (a), false, (b), (short)0, (c), false, false)

// Load a 16x32 bf16 fragment (A-layout; identical pattern serves B = W row-major
// since gemm is X @ W^T).  Lane L holds row (L&15); k-offsets per ISA 7.12.2:
// lanes 0-15: K {0..7,16..23}, lanes 16-31: K {8..15,24..31}.
__device__ __forceinline__ v16bf load_frag(const __bf16* p0, int ld) {
    const int lane = threadIdx.x & 31;
    const __bf16* p = p0 + (size_t)(lane & 15) * ld + ((lane >> 4) << 3);
    v8bf lo = *(const v8bf*)(p);
    v8bf hi = *(const v8bf*)(p + 16);
    return __builtin_shufflevector(lo, hi, 0,1,2,3,4,5,6,7,8,9,10,11,12,13,14,15);
}

__device__ __forceinline__ float sigmoidf_(float x) { return 1.f / (1.f + expf(-x)); }
__device__ __forceinline__ float siluf_(float x)    { return x / (1.f + expf(-x)); }

// ---------------- weight f32 -> bf16 (with optional K zero-pad) ----------------
__global__ void k_cvt(const float* __restrict__ src, __bf16* __restrict__ dst,
                      int rows, int sc, int dc) {
    size_t i = (size_t)blockIdx.x * 256 + threadIdx.x;
    size_t total = (size_t)rows * dc;
    if (i >= total) return;
    int r = (int)(i / dc), c = (int)(i % dc);
    dst[i] = (__bf16)((c < sc) ? src[(size_t)r * sc + c] : 0.f);
}

// ---------------- GEMM(16 x KPAD x 512) + LayerNorm + SiLU -> bf16 -------------
// mode 0: A = cat(stoch[B,1024], act[B,60], pad4)  KPAD=1088
// mode 1: A = cat(deter_bf16[B,512], obs_f32[B,5120]) KPAD=5632
template <int MODE, int KPAD>
__global__ __launch_bounds__(256) void k_lin_ln_silu(
    const float* __restrict__ srcA0, const float* __restrict__ srcA1,
    const __bf16* __restrict__ srcAbf,
    const __bf16* __restrict__ Wbf, const float* __restrict__ g,
    const float* __restrict__ b, __bf16* __restrict__ xout) {
    __shared__ __bf16 Abuf[16][64];
    __shared__ float  accS[16][512];
    __shared__ float  red[2][16][16];
    __shared__ float  stat[2][16];

    const int tid = threadIdx.x;
    const int wave = tid >> 5, lane = tid & 31;
    const int row0 = blockIdx.x * 16;

    v8f acc0 = {}, acc1 = {}, acc2 = {}, acc3 = {};
    const __bf16* Wb = Wbf + (size_t)(wave * 64) * KPAD;

    // software pipeline: B-fragment global loads run one 32-step ahead,
    // carried across the staging barrier (they don't touch LDS).
    v16bf b0 = load_frag(Wb + (size_t)0  * KPAD, KPAD);
    v16bf b1 = load_frag(Wb + (size_t)16 * KPAD, KPAD);
    v16bf b2 = load_frag(Wb + (size_t)32 * KPAD, KPAD);
    v16bf b3 = load_frag(Wb + (size_t)48 * KPAD, KPAD);

    for (int kc = 0; kc < KPAD; kc += 64) {
        #pragma unroll
        for (int i = tid; i < 16 * 64; i += 256) {
            int r = i >> 6, c = i & 63;
            int k = kc + c;
            __bf16 v;
            if (MODE == 0) {
                float f;
                if (k < 1024)      f = srcA0[(size_t)(row0 + r) * 1024 + k];
                else if (k < 1084) f = srcA1[(size_t)(row0 + r) * 60 + (k - 1024)];
                else               f = 0.f;
                v = (__bf16)f;
            } else {
                if (k < 512) v = srcAbf[(size_t)(row0 + r) * 512 + k];
                else         v = (__bf16)srcA0[(size_t)(row0 + r) * 5120 + (k - 512)];
            }
            Abuf[r][c] = v;
        }
        __syncthreads();
        #pragma unroll
        for (int ks = 0; ks < 2; ++ks) {
            int kk = kc + ks * 32;
            int kn = (kk + 32 < KPAD) ? kk + 32 : kk;   // clamped prefetch index
            v16bf n0 = load_frag(Wb + (size_t)0  * KPAD + kn, KPAD);
            v16bf n1 = load_frag(Wb + (size_t)16 * KPAD + kn, KPAD);
            v16bf n2 = load_frag(Wb + (size_t)32 * KPAD + kn, KPAD);
            v16bf n3 = load_frag(Wb + (size_t)48 * KPAD + kn, KPAD);
            v16bf a  = load_frag(&Abuf[0][ks * 32], 64);
            acc0 = WMMA_BF16(a, b0, acc0);
            acc1 = WMMA_BF16(a, b1, acc1);
            acc2 = WMMA_BF16(a, b2, acc2);
            acc3 = WMMA_BF16(a, b3, acc3);
            b0 = n0; b1 = n1; b2 = n2; b3 = n3;
        }
        __syncthreads();
    }

    const int nl = lane & 15, mh = (lane >> 4) * 8;
    #pragma unroll
    for (int v = 0; v < 8; ++v) {
        accS[mh + v][wave * 64 +  0 + nl] = acc0[v];
        accS[mh + v][wave * 64 + 16 + nl] = acc1[v];
        accS[mh + v][wave * 64 + 32 + nl] = acc2[v];
        accS[mh + v][wave * 64 + 48 + nl] = acc3[v];
    }
    __syncthreads();

    {   // per-row mean / var (two-level reduction)
        int r = tid & 15, seg = tid >> 4;
        float s = 0.f, s2 = 0.f;
        #pragma unroll 8
        for (int i = 0; i < 32; ++i) {
            float v = accS[r][seg * 32 + i];
            s += v; s2 += v * v;
        }
        red[0][r][seg] = s; red[1][r][seg] = s2;
    }
    __syncthreads();
    if (tid < 16) {
        float s = 0.f, s2 = 0.f;
        #pragma unroll
        for (int i = 0; i < 16; ++i) { s += red[0][tid][i]; s2 += red[1][tid][i]; }
        float mean = s * (1.f / 512.f);
        float var  = s2 * (1.f / 512.f) - mean * mean;
        stat[0][tid] = mean;
        stat[1][tid] = rsqrtf(var + 1e-3f);
    }
    __syncthreads();
    {
        int r = tid >> 4, c0 = (tid & 15) * 32;
        float mean = stat[0][r], rstd = stat[1][r];
        #pragma unroll 8
        for (int i = 0; i < 32; ++i) {
            int c = c0 + i;
            float v = (accS[r][c] - mean) * rstd * g[c] + b[c];
            xout[(size_t)(row0 + r) * 512 + c] = (__bf16)siluf_(v);
        }
    }
}

// ---------------- fused GRU: gi = x@Wih^T, gh = h@Whh^T, gates, new deter ------
__global__ __launch_bounds__(256) void k_gru(
    const __bf16* __restrict__ xbf, const float* __restrict__ hin,
    const __bf16* __restrict__ Wih, const __bf16* __restrict__ Whh,
    const float* __restrict__ bih, const float* __restrict__ bhh,
    float* __restrict__ out, __bf16* __restrict__ dbf) {
    __shared__ __bf16 Xs[16][512];
    __shared__ __bf16 Hs[16][512];
    const int tid = threadIdx.x, wave = tid >> 5, lane = tid & 31;
    const int row0 = blockIdx.x * 16;

    for (int i = tid; i < 16 * 512; i += 256) {
        int r = i >> 9, c = i & 511;
        Xs[r][c] = xbf[(size_t)(row0 + r) * 512 + c];
        Hs[r][c] = (__bf16)hin[(size_t)(row0 + r) * 512 + c];
    }
    __syncthreads();

    const int n0 = blockIdx.y * 128 + wave * 16;
    const __bf16* Wr = Wih + (size_t)(n0)         * 512;
    const __bf16* Wz = Wih + (size_t)(n0 + 512)  * 512;
    const __bf16* Wn = Wih + (size_t)(n0 + 1024) * 512;
    const __bf16* Vr = Whh + (size_t)(n0)         * 512;
    const __bf16* Vz = Whh + (size_t)(n0 + 512)  * 512;
    const __bf16* Vn = Whh + (size_t)(n0 + 1024) * 512;

    v8f gir = {}, giz = {}, gin = {}, ghr = {}, ghz = {}, ghn = {};
    // software pipeline: prefetch k+32 fragments before issuing k's WMMAs
    v16bf ax = load_frag(&Xs[0][0], 512);
    v16bf ah = load_frag(&Hs[0][0], 512);
    v16bf b0 = load_frag(Wr, 512);
    v16bf b1 = load_frag(Wz, 512);
    v16bf b2 = load_frag(Wn, 512);
    v16bf b3 = load_frag(Vr, 512);
    v16bf b4 = load_frag(Vz, 512);
    v16bf b5 = load_frag(Vn, 512);
    #pragma unroll 4
    for (int k = 0; k < 512; k += 32) {
        const int kn = (k + 32 < 512) ? k + 32 : k;
        v16bf nax = load_frag(&Xs[0][kn], 512);
        v16bf nah = load_frag(&Hs[0][kn], 512);
        v16bf nb0 = load_frag(Wr + kn, 512);
        v16bf nb1 = load_frag(Wz + kn, 512);
        v16bf nb2 = load_frag(Wn + kn, 512);
        v16bf nb3 = load_frag(Vr + kn, 512);
        v16bf nb4 = load_frag(Vz + kn, 512);
        v16bf nb5 = load_frag(Vn + kn, 512);
        gir = WMMA_BF16(ax, b0, gir);
        giz = WMMA_BF16(ax, b1, giz);
        gin = WMMA_BF16(ax, b2, gin);
        ghr = WMMA_BF16(ah, b3, ghr);
        ghz = WMMA_BF16(ah, b4, ghz);
        ghn = WMMA_BF16(ah, b5, ghn);
        ax = nax; ah = nah;
        b0 = nb0; b1 = nb1; b2 = nb2; b3 = nb3; b4 = nb4; b5 = nb5;
    }

    const int n = n0 + (lane & 15);
    const float bir = bih[n], biz = bih[n + 512], bin_ = bih[n + 1024];
    const float bhr = bhh[n], bhz = bhh[n + 512], bhn = bhh[n + 1024];
    const int mh = (lane >> 4) * 8;
    #pragma unroll
    for (int v = 0; v < 8; ++v) {
        int row = row0 + mh + v;
        float h  = hin[(size_t)row * 512 + n];
        float r  = sigmoidf_((gir[v] + bir) + (ghr[v] + bhr));
        float z  = sigmoidf_((giz[v] + biz) + (ghz[v] + bhz));
        float ng = tanhf((gin[v] + bin_) + r * (ghn[v] + bhn));
        float d  = (1.f - z) * ng + z * h;
        out[(size_t)row * 2560 + n] = d;
        dbf[(size_t)row * 512 + n] = (__bf16)d;
    }
}

// ---------------- logits = y@Wst^T + b; unimix softmax; argmax onehot ----------
__global__ __launch_bounds__(256) void k_stats(
    const __bf16* __restrict__ ybf, const __bf16* __restrict__ Wst,
    const float* __restrict__ bst, float* __restrict__ out) {
    __shared__ __bf16 Ys[16][512];
    const int tid = threadIdx.x, wave = tid >> 5, lane = tid & 31;
    const int row0 = blockIdx.x * 16;

    for (int i = tid; i < 16 * 512; i += 256) {
        int r = i >> 9, c = i & 511;
        Ys[r][c] = ybf[(size_t)(row0 + r) * 512 + c];
    }
    __syncthreads();

    const __bf16* Wb = Wst + (size_t)(wave * 128) * 512;
    v8f a0 = {}, a1 = {}, a2 = {}, a3 = {}, a4 = {}, a5 = {}, a6 = {}, a7 = {};
    v16bf a  = load_frag(&Ys[0][0], 512);
    v16bf b0 = load_frag(Wb + (size_t)(0 * 16) * 512, 512);
    v16bf b1 = load_frag(Wb + (size_t)(1 * 16) * 512, 512);
    v16bf b2 = load_frag(Wb + (size_t)(2 * 16) * 512, 512);
    v16bf b3 = load_frag(Wb + (size_t)(3 * 16) * 512, 512);
    v16bf b4 = load_frag(Wb + (size_t)(4 * 16) * 512, 512);
    v16bf b5 = load_frag(Wb + (size_t)(5 * 16) * 512, 512);
    v16bf b6 = load_frag(Wb + (size_t)(6 * 16) * 512, 512);
    v16bf b7 = load_frag(Wb + (size_t)(7 * 16) * 512, 512);
    #pragma unroll 4
    for (int k = 0; k < 512; k += 32) {
        const int kn = (k + 32 < 512) ? k + 32 : k;
        v16bf na  = load_frag(&Ys[0][kn], 512);
        v16bf nb0 = load_frag(Wb + (size_t)(0 * 16) * 512 + kn, 512);
        v16bf nb1 = load_frag(Wb + (size_t)(1 * 16) * 512 + kn, 512);
        v16bf nb2 = load_frag(Wb + (size_t)(2 * 16) * 512 + kn, 512);
        v16bf nb3 = load_frag(Wb + (size_t)(3 * 16) * 512 + kn, 512);
        v16bf nb4 = load_frag(Wb + (size_t)(4 * 16) * 512 + kn, 512);
        v16bf nb5 = load_frag(Wb + (size_t)(5 * 16) * 512 + kn, 512);
        v16bf nb6 = load_frag(Wb + (size_t)(6 * 16) * 512 + kn, 512);
        v16bf nb7 = load_frag(Wb + (size_t)(7 * 16) * 512 + kn, 512);
        a0 = WMMA_BF16(a, b0, a0);
        a1 = WMMA_BF16(a, b1, a1);
        a2 = WMMA_BF16(a, b2, a2);
        a3 = WMMA_BF16(a, b3, a3);
        a4 = WMMA_BF16(a, b4, a4);
        a5 = WMMA_BF16(a, b5, a5);
        a6 = WMMA_BF16(a, b6, a6);
        a7 = WMMA_BF16(a, b7, a7);
        a = na;
        b0 = nb0; b1 = nb1; b2 = nb2; b3 = nb3;
        b4 = nb4; b5 = nb5; b6 = nb6; b7 = nb7;
    }
    // write logits (+bias) straight into d_out[:, 512:1536]
    const int nl = lane & 15, mh = (lane >> 4) * 8;
    v8f* accs[8] = { &a0, &a1, &a2, &a3, &a4, &a5, &a6, &a7 };
    #pragma unroll
    for (int t = 0; t < 8; ++t) {
        int n = wave * 128 + t * 16 + nl;
        float bias = bst[n];
        #pragma unroll
        for (int v = 0; v < 8; ++v)
            out[(size_t)(row0 + mh + v) * 2560 + 512 + n] = (*accs[t])[v] + bias;
    }
    __syncthreads();  // block-level fence; logits visible within workgroup

    // 16 rows * 32 groups = 512 softmax groups, 2 per thread
    for (int gidx = tid; gidx < 512; gidx += 256) {
        int r = gidx >> 5, grp = gidx & 31;
        size_t base = (size_t)(row0 + r) * 2560;
        const float* lg = out + base + 512 + grp * 32;
        float mx = -INFINITY; int am = 0;
        #pragma unroll 8
        for (int j = 0; j < 32; ++j)
            if (lg[j] > mx) { mx = lg[j]; am = j; }
        float s = 0.f;
        #pragma unroll 8
        for (int j = 0; j < 32; ++j) s += expf(lg[j] - mx);
        float inv = 1.f / s;
        #pragma unroll 8
        for (int j = 0; j < 32; ++j) {
            float prob = 0.99f * (expf(lg[j] - mx) * inv) + (0.01f / 32.f);
            float oh = (j == am) ? 1.f : 0.f;
            // straight-through: onehot + probs - stop_grad(probs)
            out[base + 1536 + grp * 32 + j] = (oh + prob) - prob;
        }
    }
}

extern "C" void kernel_launch(void* const* d_in, const int* in_sizes, int n_in,
                              void* d_out, int out_size, void* d_ws, size_t ws_size,
                              hipStream_t stream) {
    const float* obs   = (const float*)d_in[0];
    const float* act   = (const float*)d_in[1];
    const float* stoch = (const float*)d_in[2];
    const float* deter = (const float*)d_in[3];
    const float* W_in  = (const float*)d_in[4];
    const float* g_in  = (const float*)d_in[5];
    const float* b_in  = (const float*)d_in[6];
    const float* W_ih  = (const float*)d_in[7];
    const float* W_hh  = (const float*)d_in[8];
    const float* b_ih  = (const float*)d_in[9];
    const float* b_hh  = (const float*)d_in[10];
    const float* W_ob  = (const float*)d_in[11];
    const float* g_ob  = (const float*)d_in[12];
    const float* b_ob  = (const float*)d_in[13];
    const float* W_st  = (const float*)d_in[14];
    const float* b_st  = (const float*)d_in[15];

    const int B = in_sizes[0] / 5120;           // 16384
    float* out = (float*)d_out;

    char* ws = (char*)d_ws;
    size_t off = 0;
    auto alloc = [&](size_t bytes) -> void* {
        void* p = ws + off;
        off += (bytes + 255) & ~(size_t)255;
        return p;
    };
    __bf16* Winbf = (__bf16*)alloc((size_t)512 * 1088 * 2);
    __bf16* Wihbf = (__bf16*)alloc((size_t)1536 * 512 * 2);
    __bf16* Whhbf = (__bf16*)alloc((size_t)1536 * 512 * 2);
    __bf16* Wobbf = (__bf16*)alloc((size_t)512 * 5632 * 2);
    __bf16* Wstbf = (__bf16*)alloc((size_t)1024 * 512 * 2);
    __bf16* xbf   = (__bf16*)alloc((size_t)B * 512 * 2);
    __bf16* dbf   = (__bf16*)alloc((size_t)B * 512 * 2);
    __bf16* ybf   = (__bf16*)alloc((size_t)B * 512 * 2);
    if (off > ws_size) return;  // not enough scratch; bail deterministically

    auto blocks = [](size_t n) { return (unsigned)((n + 255) / 256); };
    k_cvt<<<blocks((size_t)512 * 1088), 256, 0, stream>>>(W_in, Winbf, 512, 1084, 1088);
    k_cvt<<<blocks((size_t)1536 * 512), 256, 0, stream>>>(W_ih, Wihbf, 1536, 512, 512);
    k_cvt<<<blocks((size_t)1536 * 512), 256, 0, stream>>>(W_hh, Whhbf, 1536, 512, 512);
    k_cvt<<<blocks((size_t)512 * 5632), 256, 0, stream>>>(W_ob, Wobbf, 512, 5632, 5632);
    k_cvt<<<blocks((size_t)1024 * 512), 256, 0, stream>>>(W_st, Wstbf, 1024, 512, 512);

    const int rb = B / 16;  // row blocks of 16
    k_lin_ln_silu<0, 1088><<<rb, 256, 0, stream>>>(stoch, act, nullptr,
                                                   Winbf, g_in, b_in, xbf);
    k_gru<<<dim3(rb, 4), 256, 0, stream>>>(xbf, deter, Wihbf, Whhbf,
                                           b_ih, b_hh, out, dbf);
    k_lin_ln_silu<1, 5632><<<rb, 256, 0, stream>>>(obs, nullptr, dbf,
                                                   Wobbf, g_ob, b_ob, ybf);
    k_stats<<<rb, 256, 0, stream>>>(ybf, Wstbf, b_st, out);
}
